// Correlation3D_83786222010961
// MI455X (gfx1250) — matrix operand order repaired and checked by reading, not compiled, not run on hardware
//
#include <hip/hip_runtime.h>
#include <hip/hip_bf16.h>

// ---------------------------------------------------------------------------
// CamLiFlow-style Correlation3D for MI455X (gfx1250, wave32, WMMA).
// Pipeline:
//   prep:   xyz -> [B,N,4] f32 ; feat -> [B,N,64] f16 ; weights -> A-fragments
//   knn:    brute force top-16 (LDS point tiles, register insertion sort)
//   stage1: per-wave query point, v_wmma_f32_16x16x32_f16 MLP (131->64->64),
//           weight-net (3->8->8 VALU, 8->64 WMMA), product + shuffle reduce
//   stage2: weight-net WMMA + gather of stage1 output, product + reduce
// ---------------------------------------------------------------------------

typedef __attribute__((ext_vector_type(16))) _Float16 v16h;
typedef __attribute__((ext_vector_type(8)))  _Float16 v8h;
typedef __attribute__((ext_vector_type(8)))  float    v8f;

#define KNN_K 16
#define LEAKY 0.1f

__device__ inline v8f zero8f() {
  v8f z;
#pragma unroll
  for (int i = 0; i < 8; ++i) z[i] = 0.f;
  return z;
}

__device__ inline v16h zero16h() {
  v16h z;
#pragma unroll
  for (int i = 0; i < 16; ++i) z[i] = (_Float16)0.f;
  return z;
}

__device__ inline v8f wmma_f16(v16h a, v16h b, v8f c) {
  // D = A(16x32 f16) * B(32x16 f16) + C(16x16 f32)
  return __builtin_amdgcn_wmma_f32_16x16x32_f16(
      false, a, false, b, (short)0, c, false, false);
}

// A-fragment buffers: [tile][lane][16] f16, tile = mt*KC + kc
__device__ inline v16h ldA(const _Float16* base, int tile, int lane) {
  return *reinterpret_cast<const v16h*>(base + ((size_t)tile * 32 + lane) * 16);
}

// ---------------------------------------------------------------------------
// Prep: layout transforms
// ---------------------------------------------------------------------------

// [B,64,N] f32 -> [B,N,64] f16 (point-major: neighbor gather = contiguous 128B)
__global__ __launch_bounds__(256) void k_feat_t(const float* __restrict__ in,
                                                _Float16* __restrict__ out, int N) {
  int t = blockIdx.x * 256 + threadIdx.x;   // over B*64*N
  int n = t % N;
  int c = (t / N) & 63;
  int b = t / (N * 64);
  out[((size_t)b * N + n) * 64 + c] = (_Float16)in[t];
}

// [B,3,N] f32 -> [B,N,4] f32
__global__ __launch_bounds__(256) void k_xyz_t(const float* __restrict__ in,
                                               float* __restrict__ out, int N) {
  int t = blockIdx.x * 256 + threadIdx.x;   // over B*3*N
  int n = t % N;
  int c = (t / N) % 3;
  int b = t / (N * 3);
  out[((size_t)b * N + n) * 4 + c] = in[t];
}

// Swizzle weight matrix W[out , in=IN] into CDNA5 16x32 f16 A-fragments.
// A layout used here (contiguous k, same as B operands, not ISA table order): lane 0-15 row M=lane, lane 16-31 same rows;
//  element j: K = j (+16 for upper half-lanes)
__device__ inline void fill_frag(_Float16* dst, const float* W, int IN, int KC, int t) {
  int j    = t & 15;
  int lane = (t >> 4) & 31;
  int tile = t >> 9;
  int mt = tile / KC, kc = tile % KC;
  int o    = mt * 16 + (lane & 15);
  int half = lane >> 4;
  int r = j >> 1;
  int kl = half * 16 + j; (void)r;
  int c = kc * 32 + kl;
  dst[t] = (c < IN) ? (_Float16)W[o * IN + c] : (_Float16)0.f;
}

__global__ __launch_bounds__(256) void k_prep_weights(
    const float* __restrict__ w1, const float* __restrict__ w2,
    const float* __restrict__ wn2w3, const float* __restrict__ wn1w3,
    _Float16* __restrict__ f1, _Float16* __restrict__ f2,
    _Float16* __restrict__ fn2, _Float16* __restrict__ fn1) {
  int t = blockIdx.x * 256 + threadIdx.x;
  if      (t < 10240) fill_frag(f1,  w1,    131, 5, t);          // 4x5 tiles
  else if (t < 14336) fill_frag(f2,  w2,     64, 2, t - 10240);  // 4x2 tiles
  else if (t < 16384) fill_frag(fn2, wn2w3,   8, 1, t - 14336);  // 4x1 tiles
  else if (t < 18432) fill_frag(fn1, wn1w3,   8, 1, t - 16384);  // 4x1 tiles
}

// ---------------------------------------------------------------------------
// Brute-force KNN, one thread per query, LDS point tiles, register top-16.
// ---------------------------------------------------------------------------
__global__ __launch_bounds__(256) void k_knn(const float* __restrict__ qxyz,   // [B,N,4]
                                             const float* __restrict__ pxyz,   // [B,N,4]
                                             int* __restrict__ idx, int N) {
  __shared__ float4 tile[512];
  int t = blockIdx.x * 256 + threadIdx.x;    // query = b*N+n (N % 256 == 0)
  int b = t / N;
  float qx = qxyz[(size_t)t * 4 + 0];
  float qy = qxyz[(size_t)t * 4 + 1];
  float qz = qxyz[(size_t)t * 4 + 2];

  float bestd[KNN_K];
  int   besti[KNN_K];
#pragma unroll
  for (int j = 0; j < KNN_K; ++j) { bestd[j] = 3.402823466e38f; besti[j] = 0; }

  const float4* pb = reinterpret_cast<const float4*>(pxyz) + (size_t)b * N;
  for (int t0 = 0; t0 < N; t0 += 512) {
    __syncthreads();
#pragma unroll
    for (int i = 0; i < 2; ++i)
      tile[threadIdx.x + i * 256] = pb[t0 + threadIdx.x + i * 256];
    __syncthreads();
    for (int j = 0; j < 512; ++j) {
      float4 p = tile[j];
      float dx = p.x - qx, dy = p.y - qy, dz = p.z - qz;
      float d2 = dx * dx + dy * dy + dz * dz;
      if (d2 < bestd[KNN_K - 1]) {      // rare after warm-up
        float d = d2; int id = t0 + j;
#pragma unroll
        for (int s = 0; s < KNN_K; ++s) {
          bool sw = d < bestd[s];
          float td = sw ? bestd[s] : d;  int ti = sw ? besti[s] : id;
          bestd[s] = sw ? d : bestd[s];  besti[s] = sw ? id : besti[s];
          d = td; id = ti;
        }
      }
    }
  }
#pragma unroll
  for (int j = 0; j < KNN_K; ++j) idx[(size_t)t * KNN_K + j] = besti[j];
}

// ---------------------------------------------------------------------------
// Stage 1: one wave per query point (b,n).  16 neighbors = 16 WMMA columns.
// ---------------------------------------------------------------------------
__global__ __launch_bounds__(256) void k_stage1(
    const int* __restrict__ idx12,
    const float* __restrict__ xyz1t, const float* __restrict__ xyz2t,
    const _Float16* __restrict__ feat1t, const _Float16* __restrict__ feat2t,
    const _Float16* __restrict__ fragW1, const _Float16* __restrict__ fragW2,
    const _Float16* __restrict__ fragWn3,
    const float* __restrict__ b1, const float* __restrict__ b2,
    const float* __restrict__ wn_w1, const float* __restrict__ wn_b1,
    const float* __restrict__ wn_w2, const float* __restrict__ wn_b2,
    const float* __restrict__ wn_b3,
    float* __restrict__ p2nt, int N) {
  __shared__ _Float16 hsh[8][16][64];     // per-wave h1, column-major f16
  const int lane = threadIdx.x & 31;
  const int wid  = threadIdx.x >> 5;
  const int q    = blockIdx.x * 8 + wid;  // == b*N + n, grid sized exactly
  const int b    = q / N;
  const int col  = lane & 15;             // neighbor index
  const int half = lane >> 4;

  const int m     = idx12[(size_t)q * KNN_K + col];
  const size_t pb = (size_t)b * N + m;

  const float qx = xyz1t[(size_t)q * 4 + 0];
  const float qy = xyz1t[(size_t)q * 4 + 1];
  const float qz = xyz1t[(size_t)q * 4 + 2];
  const float dx = xyz2t[pb * 4 + 0] - qx;
  const float dy = xyz2t[pb * 4 + 1] - qy;
  const float dz = xyz2t[pb * 4 + 2] - qz;

  // ---- layer1: cat(feat1, knn_feat2, dxyz) [131 -> pad 160] x 16 nbrs ----
  v8f acc[4];
#pragma unroll
  for (int mt = 0; mt < 4; ++mt) acc[mt] = zero8f();
#pragma unroll
  for (int kc = 0; kc < 5; ++kc) {
    v16h bf;
    if (kc < 2) {         // feat1 (same for every neighbor column)
      bf = *reinterpret_cast<const v16h*>(feat1t + (size_t)q * 64 + kc * 32 + half * 16);
    } else if (kc < 4) {  // gathered feat2
      bf = *reinterpret_cast<const v16h*>(feat2t + pb * 64 + (kc - 2) * 32 + half * 16);
    } else {              // dxyz + zero pad
      bf = zero16h();
      bf[0] = half == 0 ? (_Float16)dx : (_Float16)0.f;
      bf[1] = half == 0 ? (_Float16)dy : (_Float16)0.f;
      bf[2] = half == 0 ? (_Float16)dz : (_Float16)0.f;
    }
#pragma unroll
    for (int mt = 0; mt < 4; ++mt)
      acc[mt] = wmma_f16(ldA(fragW1, mt * 5 + kc, lane), bf, acc[mt]);
  }

  // bias + LeakyReLU, then stash h1 into LDS as [col][ch] f16
#pragma unroll
  for (int mt = 0; mt < 4; ++mt) {
    v8h hh;
#pragma unroll
    for (int r = 0; r < 8; ++r) {
      int ch = mt * 16 + half * 8 + r;         // D layout: row = r + 8*half
      float v = acc[mt][r] + b1[ch];
      hh[r] = (_Float16)(v > 0.f ? v : LEAKY * v);
    }
    *reinterpret_cast<v8h*>(&hsh[wid][col][mt * 16 + half * 8]) = hh;
  }
  __syncthreads();

  // ---- layer2: 64 -> 64 ----
  v8f acc2[4];
#pragma unroll
  for (int mt = 0; mt < 4; ++mt) acc2[mt] = zero8f();
#pragma unroll
  for (int kc = 0; kc < 2; ++kc) {
    v16h bf = *reinterpret_cast<const v16h*>(&hsh[wid][col][kc * 32 + half * 16]);
#pragma unroll
    for (int mt = 0; mt < 4; ++mt)
      acc2[mt] = wmma_f16(ldA(fragW2, mt * 2 + kc, lane), bf, acc2[mt]);
  }
  v8f h2[4];
#pragma unroll
  for (int mt = 0; mt < 4; ++mt)
#pragma unroll
    for (int r = 0; r < 8; ++r) {
      float v = acc2[mt][r] + b2[mt * 16 + half * 8 + r];
      h2[mt][r] = v > 0.f ? v : LEAKY * v;
    }

  // ---- weight-net: 3 -> 8 -> 8 on VALU, 8 -> 64 via WMMA ----
  float h0[8], hv[8];
#pragma unroll
  for (int o = 0; o < 8; ++o) {
    float s = wn_b1[o] + wn_w1[o * 3 + 0] * dx + wn_w1[o * 3 + 1] * dy + wn_w1[o * 3 + 2] * dz;
    h0[o] = fmaxf(s, 0.f);
  }
#pragma unroll
  for (int o = 0; o < 8; ++o) {
    float s = wn_b2[o];
#pragma unroll
    for (int c = 0; c < 8; ++c) s += wn_w2[o * 8 + c] * h0[c];
    hv[o] = fmaxf(s, 0.f);
  }
  v16h bw = zero16h();
#pragma unroll
  for (int j = 0; j < 8; ++j)
    bw[j] = half == 0 ? (_Float16)hv[j] : (_Float16)0.f;   // K=0..7 live in lanes 0-15
  v8f accw[4];
#pragma unroll
  for (int mt = 0; mt < 4; ++mt)
    accw[mt] = wmma_f16(ldA(fragWn3, mt, lane), bw, zero8f());

  // ---- product + column reduction (sum over the 16 neighbors) ----
  float red[4][8];
#pragma unroll
  for (int mt = 0; mt < 4; ++mt)
#pragma unroll
    for (int r = 0; r < 8; ++r) {
      float w = fmaxf(accw[mt][r] + wn_b3[mt * 16 + half * 8 + r], 0.f);
      float p = w * h2[mt][r];
      p += __shfl_xor(p, 1, 32);
      p += __shfl_xor(p, 2, 32);
      p += __shfl_xor(p, 4, 32);
      p += __shfl_xor(p, 8, 32);   // stays within the 16-lane half
      red[mt][r] = p;
    }
  if (col == 0) {
#pragma unroll
    for (int mt = 0; mt < 4; ++mt)
#pragma unroll
      for (int r = 0; r < 8; ++r)
        p2nt[(size_t)q * 64 + mt * 16 + half * 8 + r] = red[mt][r];
  }
}

// ---------------------------------------------------------------------------
// Stage 2: weight-net (wn1) WMMA + gather of p2n, product + reduce.
// ---------------------------------------------------------------------------
__global__ __launch_bounds__(256) void k_stage2(
    const int* __restrict__ idx11, const float* __restrict__ xyz1t,
    const _Float16* __restrict__ fragWn3,
    const float* __restrict__ wn_w1, const float* __restrict__ wn_b1,
    const float* __restrict__ wn_w2, const float* __restrict__ wn_b2,
    const float* __restrict__ wn_b3,
    const float* __restrict__ p2nt, float* __restrict__ out, int N) {
  const int lane = threadIdx.x & 31;
  const int wid  = threadIdx.x >> 5;
  const int q    = blockIdx.x * 8 + wid;
  const int b    = q / N;
  const int n    = q - b * N;
  const int col  = lane & 15;
  const int half = lane >> 4;

  const int m     = idx11[(size_t)q * KNN_K + col];
  const size_t pb = (size_t)b * N + m;
  const float dx = xyz1t[pb * 4 + 0] - xyz1t[(size_t)q * 4 + 0];
  const float dy = xyz1t[pb * 4 + 1] - xyz1t[(size_t)q * 4 + 1];
  const float dz = xyz1t[pb * 4 + 2] - xyz1t[(size_t)q * 4 + 2];

  float h0[8], hv[8];
#pragma unroll
  for (int o = 0; o < 8; ++o) {
    float s = wn_b1[o] + wn_w1[o * 3 + 0] * dx + wn_w1[o * 3 + 1] * dy + wn_w1[o * 3 + 2] * dz;
    h0[o] = fmaxf(s, 0.f);
  }
#pragma unroll
  for (int o = 0; o < 8; ++o) {
    float s = wn_b2[o];
#pragma unroll
    for (int c = 0; c < 8; ++c) s += wn_w2[o * 8 + c] * h0[c];
    hv[o] = fmaxf(s, 0.f);
  }
  v16h bw = zero16h();
#pragma unroll
  for (int j = 0; j < 8; ++j)
    bw[j] = half == 0 ? (_Float16)hv[j] : (_Float16)0.f;

  v8f accw[4];
#pragma unroll
  for (int mt = 0; mt < 4; ++mt)
    accw[mt] = wmma_f16(ldA(fragWn3, mt, lane), bw, zero8f());

  float red[4][8];
#pragma unroll
  for (int mt = 0; mt < 4; ++mt) {
    // gather 8 consecutive channels of neighbor m in D-fragment layout
    const float4* cp = reinterpret_cast<const float4*>(p2nt + pb * 64 + mt * 16 + half * 8);
    float4 c0 = cp[0], c1 = cp[1];
    float cost[8] = {c0.x, c0.y, c0.z, c0.w, c1.x, c1.y, c1.z, c1.w};
#pragma unroll
    for (int r = 0; r < 8; ++r) {
      float w = fmaxf(accw[mt][r] + wn_b3[mt * 16 + half * 8 + r], 0.f);
      float p = w * cost[r];
      p += __shfl_xor(p, 1, 32);
      p += __shfl_xor(p, 2, 32);
      p += __shfl_xor(p, 4, 32);
      p += __shfl_xor(p, 8, 32);
      red[mt][r] = p;
    }
  }
  if (col == 0) {
#pragma unroll
    for (int mt = 0; mt < 4; ++mt)
#pragma unroll
      for (int r = 0; r < 8; ++r) {
        int ch = mt * 16 + half * 8 + r;
        out[((size_t)b * 64 + ch) * N + n] = red[mt][r];   // [B,64,N]
      }
  }
}

// ---------------------------------------------------------------------------
extern "C" void kernel_launch(void* const* d_in, const int* in_sizes, int n_in,
                              void* d_out, int out_size, void* d_ws, size_t ws_size,
                              hipStream_t stream) {
  const float* xyz1   = (const float*)d_in[0];
  const float* feat1  = (const float*)d_in[1];
  const float* xyz2   = (const float*)d_in[2];
  const float* feat2  = (const float*)d_in[3];
  const float* cost_w1 = (const float*)d_in[4];
  const float* cost_b1 = (const float*)d_in[5];
  const float* cost_w2 = (const float*)d_in[6];
  const float* cost_b2 = (const float*)d_in[7];
  const float* wn1_w1 = (const float*)d_in[8];
  const float* wn1_b1 = (const float*)d_in[9];
  const float* wn1_w2 = (const float*)d_in[10];
  const float* wn1_b2 = (const float*)d_in[11];
  const float* wn1_w3 = (const float*)d_in[12];
  const float* wn1_b3 = (const float*)d_in[13];
  const float* wn2_w1 = (const float*)d_in[14];
  const float* wn2_b1 = (const float*)d_in[15];
  const float* wn2_w2 = (const float*)d_in[16];
  const float* wn2_b2 = (const float*)d_in[17];
  const float* wn2_w3 = (const float*)d_in[18];
  const float* wn2_b3 = (const float*)d_in[19];
  (void)n_in; (void)out_size; (void)ws_size;

  const int B = 4;
  const int N = in_sizes[0] / (3 * B);
  const size_t BN = (size_t)B * N;

  // workspace carve-out (256B aligned)
  char* ws = (char*)d_ws;
  size_t off = 0;
  auto take = [&](size_t bytes) -> void* {
    void* p = ws + off;
    off += (bytes + 255) & ~(size_t)255;
    return p;
  };
  int*       idx12   = (int*)take(BN * KNN_K * sizeof(int));
  int*       idx11   = (int*)take(BN * KNN_K * sizeof(int));
  _Float16*  feat1t  = (_Float16*)take(BN * 64 * sizeof(_Float16));
  _Float16*  feat2t  = (_Float16*)take(BN * 64 * sizeof(_Float16));
  float*     xyz1t   = (float*)take(BN * 4 * sizeof(float));
  float*     xyz2t   = (float*)take(BN * 4 * sizeof(float));
  float*     p2nt    = (float*)take(BN * 64 * sizeof(float));
  _Float16*  fragW1  = (_Float16*)take(20 * 512 * sizeof(_Float16));
  _Float16*  fragW2  = (_Float16*)take(8 * 512 * sizeof(_Float16));
  _Float16*  fragWn2 = (_Float16*)take(4 * 512 * sizeof(_Float16));
  _Float16*  fragWn1 = (_Float16*)take(4 * 512 * sizeof(_Float16));

  // --- prep ---
  k_xyz_t <<<(B * 3 * N) / 256, 256, 0, stream>>>(xyz1, xyz1t, N);
  k_xyz_t <<<(B * 3 * N) / 256, 256, 0, stream>>>(xyz2, xyz2t, N);
  k_feat_t<<<(B * 64 * N) / 256, 256, 0, stream>>>(feat1, feat1t, N);
  k_feat_t<<<(B * 64 * N) / 256, 256, 0, stream>>>(feat2, feat2t, N);
  k_prep_weights<<<72, 256, 0, stream>>>(cost_w1, cost_w2, wn2_w3, wn1_w3,
                                         fragW1, fragW2, fragWn2, fragWn1);
  // --- knn ---
  k_knn<<<(int)(BN / 256), 256, 0, stream>>>(xyz1t, xyz2t, idx12, N);  // cloud1 -> cloud2
  k_knn<<<(int)(BN / 256), 256, 0, stream>>>(xyz1t, xyz1t, idx11, N);  // cloud1 -> cloud1
  // --- stage 1: p2n cost ---
  k_stage1<<<(int)(BN / 8), 256, 0, stream>>>(
      idx12, xyz1t, xyz2t, feat1t, feat2t, fragW1, fragW2, fragWn2,
      cost_b1, cost_b2, wn2_w1, wn2_b1, wn2_w2, wn2_b2, wn2_b3, p2nt, N);
  // --- stage 2: n2n cost ---
  k_stage2<<<(int)(BN / 8), 256, 0, stream>>>(
      idx11, xyz1t, fragWn1, wn1_w1, wn1_b1, wn1_w2, wn1_b2, wn1_b3,
      p2nt, (float*)d_out, N);
}
